// NeuralODENet_12773232738516
// MI455X (gfx1250) — matrix operand order repaired and checked by reading, not compiled, not run on hardware
//
#include <hip/hip_runtime.h>
#include <hip/hip_bf16.h>
#include <math.h>

// ---------------------------------------------------------------------------
// CDNA5 / gfx1250 WMMA types (wave32)
// ---------------------------------------------------------------------------
typedef __attribute__((ext_vector_type(16))) __bf16 v16bf;
typedef __attribute__((ext_vector_type(8)))  float  v8f;

union FragBF {
  v16bf v;
  uint4 q[2];
  __bf16 e[16];
};

#define BATCH 256

static __device__ __forceinline__ float  bf2f(__bf16 x) { return (float)x; }
static __device__ __forceinline__ __bf16 f2bf(float  x) { return (__bf16)x; }

// dopri5 stage-combination coefficients (row s = coeffs of k1..k6 applied after
// computing k_{s+1}; row 5 is the final state update).
__constant__ float CO[6][6] = {
  { 0.2f, 0.f, 0.f, 0.f, 0.f, 0.f },
  { 3.f/40.f, 9.f/40.f, 0.f, 0.f, 0.f, 0.f },
  { 44.f/45.f, -56.f/15.f, 32.f/9.f, 0.f, 0.f, 0.f },
  { 19372.f/6561.f, -25360.f/2187.f, 64448.f/6561.f, -212.f/729.f, 0.f, 0.f },
  { 9017.f/3168.f, -355.f/33.f, 46732.f/5247.f, 49.f/176.f, -5103.f/18656.f, 0.f },
  { 35.f/384.f, 0.f, 500.f/1113.f, 125.f/192.f, -2187.f/6784.f, 11.f/84.f },
};

// ---------------------------------------------------------------------------
// Prep kernels
// ---------------------------------------------------------------------------

__global__ void zero_kernel(__bf16* z) { z[threadIdx.x] = f2bf(0.f); }

// x [B,3,32,32] f32 -> [3][B][1024] bf16
__global__ void cvt_x_kernel(const float* __restrict__ x, __bf16* __restrict__ out) {
  int idx = blockIdx.x * 256 + threadIdx.x;
  if (idx >= 3 * BATCH * 1024) return;
  int c = idx / (BATCH * 1024);
  int r = idx - c * (BATCH * 1024);
  int b = r >> 10;
  int p = r & 1023;
  out[idx] = f2bf(x[(size_t)b * 3 * 1024 + c * 1024 + p]);
}

// conv weights [Cout][Cin][3][3] f32 -> [Cout][Kp16] bf16 where each input
// channel's 9 taps are padded to a 16-slot group (slots 9..15 = 0) and the
// channel count is padded to even.  k = g*16 + r,  g = channel, r = tap.
__global__ void cvt_convw_kernel(const float* __restrict__ w, __bf16* __restrict__ out,
                                 int Cin, int Kp16, int total) {
  int idx = blockIdx.x * 256 + threadIdx.x;
  if (idx >= total) return;
  int m = idx / Kp16;
  int k = idx - m * Kp16;
  int g = k >> 4;
  int r = k & 15;
  float v = 0.f;
  if (g < Cin && r < 9) v = w[((size_t)m * Cin + g) * 9 + r];
  out[idx] = f2bf(v);
}

// dense weight [K][N] f32 -> transposed [N][K] bf16 (contiguous WMMA B loads)
__global__ void cvt_T_kernel(const float* __restrict__ w, __bf16* __restrict__ out,
                             int K, int N) {
  int idx = blockIdx.x * 256 + threadIdx.x;
  if (idx >= K * N) return;
  int n = idx / K;
  int k = idx - n * K;
  out[idx] = f2bf(w[(size_t)k * N + n]);
}

// ---------------------------------------------------------------------------
// Implicit-GEMM 3x3 SAME conv:  y[Cout][B*H*W] = W[Cout][Kp16] x patches
// Each wave owns one 16-pixel column tile and accumulates ALL MT = Cout/16
// row tiles, so the expensive patch gather is amortized MT-fold.
// Tap index within a 16-group is a compile-time constant (no int division),
// and out-of-bounds/padded gathers read a zero page instead of branching.
// grid = (Npix/128, 1), block = 256 (8 waves)
// ---------------------------------------------------------------------------
template<int MT>
__global__ __launch_bounds__(256)
void conv_gemm_kernel(const __bf16* __restrict__ act,   // [Cin][B][H][W]
                      const __bf16* __restrict__ wbf,   // [Cout][Kp16]
                      __bf16* __restrict__ y,           // [Cout][Npix]
                      const __bf16* __restrict__ zpad,  // zero page
                      int Cin, int H, int W, int Kp16, int Npix)
{
  const int lane = threadIdx.x & 31;
  const int wave = threadIdx.x >> 5;
  const int l  = lane & 15;
  const int hi = lane >> 4;
  const int HWp = H * W;
  const int n0 = ((blockIdx.x << 3) + wave) << 4;

  const int n  = n0 + l;
  const int b  = n / HWp;
  const int p  = n - b * HWp;
  const int y0 = p / W;
  const int x0 = p - y0 * W;

  v8f acc[MT];
  #pragma unroll
  for (int mt = 0; mt < MT; ++mt)
    acc[mt] = (v8f){0.f, 0.f, 0.f, 0.f, 0.f, 0.f, 0.f, 0.f};

  const __bf16* arow = wbf + (size_t)l * Kp16;   // row l of each 16-row block

  for (int kk = 0; kk < Kp16; kk += 32) {
    // ---- B fragment: 9 unpredicated gathers (taps), rest structurally 0 ----
    const int g = (kk >> 4) + hi;                 // input-channel group
    const bool gok = g < Cin;
    const __bf16* base = act + (size_t)g * (BATCH * HWp) + (size_t)b * HWp;
    FragBF bb;
    #pragma unroll
    for (int i = 9; i < 16; ++i) bb.e[i] = f2bf(0.f);
    #pragma unroll
    for (int i = 0; i < 9; ++i) {
      const int ky = i / 3, kx = i % 3;           // compile-time constants
      int iy = y0 + ky - 1;
      int ix = x0 + kx - 1;
      bool ok = gok && (unsigned)iy < (unsigned)H && (unsigned)ix < (unsigned)W;
      const __bf16* ptr = ok ? (base + iy * W + ix) : zpad;  // no exec branch
      bb.e[i] = *ptr;
    }
    // ---- A fragments: contiguous b128 loads, one per Cout row tile ----
    #pragma unroll
    for (int mt = 0; mt < MT; ++mt) {
      const __bf16* ar = arow + (size_t)mt * 16 * Kp16 + kk + hi * 16;
      FragBF a;
      a.q[0] = *reinterpret_cast<const uint4*>(ar);
      a.q[1] = *reinterpret_cast<const uint4*>(ar + 8);
      acc[mt] = __builtin_amdgcn_wmma_f32_16x16x32_bf16(false, a.v, false, bb.v,
                                                        (short)0, acc[mt], false, false);
    }
  }
  #pragma unroll
  for (int mt = 0; mt < MT; ++mt) {
    #pragma unroll
    for (int j = 0; j < 8; ++j) {
      int m = mt * 16 + hi * 8 + j;
      y[(size_t)m * Npix + n] = f2bf(acc[mt][j]);
    }
  }
}

// ---------------------------------------------------------------------------
// BatchNorm (training-mode batch stats) on bf16 activations, per channel.
// 1024 threads/block: only C blocks exist, so per-block concurrency is what
// feeds the 23.3 TB/s HBM path on these 8-32 MB scans.
// ---------------------------------------------------------------------------
__global__ __launch_bounds__(1024)
void bn_stats_kernel(const __bf16* __restrict__ y, float* __restrict__ stats, int Npix)
{
  __shared__ float s0[1024], s1[1024];
  const int c = blockIdx.x;
  const __bf16* pc = y + (size_t)c * Npix;
  float a = 0.f, b = 0.f;
  for (int i = threadIdx.x; i < Npix; i += 1024) {
    float v = bf2f(pc[i]);
    a += v; b += v * v;
  }
  s0[threadIdx.x] = a; s1[threadIdx.x] = b;
  __syncthreads();
  for (int off = 512; off > 0; off >>= 1) {
    if (threadIdx.x < off) {
      s0[threadIdx.x] += s0[threadIdx.x + off];
      s1[threadIdx.x] += s1[threadIdx.x + off];
    }
    __syncthreads();
  }
  if (threadIdx.x == 0) {
    float mean = s0[0] / (float)Npix;
    float var  = s1[0] / (float)Npix - mean * mean;
    stats[2 * c]     = mean;
    stats[2 * c + 1] = rsqrtf(var + 1e-5f);
  }
}

__global__ void bn_apply_kernel(__bf16* __restrict__ y, const float* __restrict__ stats,
                                const float* __restrict__ g, const float* __restrict__ beta,
                                int Npix, int total)
{
  int idx = blockIdx.x * 256 + threadIdx.x;
  if (idx >= total) return;
  int c = idx / Npix;
  float v = bf2f(y[idx]);
  v = (v - stats[2 * c]) * stats[2 * c + 1] * g[c] + beta[c];
  y[idx] = f2bf(fmaxf(v, 0.f));
}

// 2x2 maxpool on [C][B][H][W] bf16
__global__ void maxpool_kernel(const __bf16* __restrict__ in, __bf16* __restrict__ out,
                               int H, int W, int total)
{
  int idx = blockIdx.x * 256 + threadIdx.x;
  if (idx >= total) return;
  int Ho = H >> 1, Wo = W >> 1;
  int per = BATCH * Ho * Wo;
  int c = idx / per;
  int r = idx - c * per;
  int b = r / (Ho * Wo);
  int p = r - b * (Ho * Wo);
  int oy = p / Wo;
  int ox = p - oy * Wo;
  const __bf16* base = in + (size_t)c * (BATCH * H * W) + (size_t)b * (H * W)
                          + (2 * oy) * W + 2 * ox;
  float v = fmaxf(fmaxf(bf2f(base[0]), bf2f(base[1])),
                  fmaxf(bf2f(base[W]), bf2f(base[W + 1])));
  out[idx] = f2bf(v);
}

// ---------------------------------------------------------------------------
// Projection GEMM: h0[B=256][256] = feats[B][4096] @ proj_w[4096][256] + b
// feats gathered from pooled act [C=256][B][16]; proj_w pre-transposed [N][K].
// grid = (2, 16), block = 256
// ---------------------------------------------------------------------------
__global__ __launch_bounds__(256)
void proj_gemm_kernel(const __bf16* __restrict__ act,   // [256][256][16]
                      const __bf16* __restrict__ wT,    // [256][4096]
                      const float* __restrict__ bias,
                      float* __restrict__ h)
{
  const int lane = threadIdx.x & 31;
  const int wave = threadIdx.x >> 5;
  const int l  = lane & 15;
  const int hi = lane >> 4;
  const int m0 = blockIdx.y << 4;
  const int n0 = ((blockIdx.x << 3) + wave) << 4;
  const int bm = m0 + l;
  const int nn = n0 + l;

  v8f acc = {0.f, 0.f, 0.f, 0.f, 0.f, 0.f, 0.f, 0.f};
  const __bf16* brow = wT + (size_t)nn * 4096;

  for (int kk = 0; kk < 4096; kk += 32) {
    FragBF a, bb;
    int c0 = kk >> 4;   // feats k = c*16 + s
    a.q[0]  = *reinterpret_cast<const uint4*>(act + (size_t)c0 * 4096 + bm * 16 + hi * 8);
    a.q[1]  = *reinterpret_cast<const uint4*>(act + (size_t)(c0 + 1) * 4096 + bm * 16 + hi * 8);
    bb.q[0] = *reinterpret_cast<const uint4*>(brow + kk + hi * 8);
    bb.q[1] = *reinterpret_cast<const uint4*>(brow + kk + hi * 8 + 16);
    acc = __builtin_amdgcn_wmma_f32_16x16x32_bf16(false, a.v, false, bb.v,
                                                  (short)0, acc, false, false);
  }
  #pragma unroll
  for (int j = 0; j < 8; ++j) {
    int m = m0 + hi * 8 + j;
    h[(size_t)m * 256 + nn] = acc[j] + bias[nn];
  }
}

// ---------------------------------------------------------------------------
// Persistent single-workgroup dopri5 integrator.
// 1024 threads = 32 waves = 8 waves per SIMD32 on one WGP: the 360 dependent
// 256^3 GEMMs are latency-bound, so we maximize per-SIMD wave occupancy to
// hide L2 weight-load latency, and each wave only computes 8 serial tiles.
// Activations ping-pong through two 128 KB LDS bf16 buffers (320 KB WGP LDS);
// the three 128 KB transposed weight matrices stay L2-resident.
// ---------------------------------------------------------------------------
__device__ __forceinline__
void gemm256(const __bf16* __restrict__ aLDS,   // [256][256] in LDS
             const __bf16* __restrict__ wT,     // [256][256] global (transposed)
             const float*  __restrict__ bias,
             __bf16* __restrict__ outLDS,       // bf16 out (or nullptr)
             float*  __restrict__ outG,         // f32 out (or nullptr)
             bool do_tanh, int wave, int lane)
{
  const int l  = lane & 15;
  const int hi = lane >> 4;
  #pragma unroll
  for (int tt = 0; tt < 8; ++tt) {              // 32 waves x 8 tiles = 256 tiles
    const int t  = wave * 8 + tt;
    const int m0 = (t >> 4) << 4;
    const int n0 = (t & 15) << 4;
    v8f acc = {0.f, 0.f, 0.f, 0.f, 0.f, 0.f, 0.f, 0.f};
    const __bf16* arow = aLDS + (size_t)(m0 + l) * 256;
    const __bf16* brow = wT   + (size_t)(n0 + l) * 256;
    #pragma unroll
    for (int kk = 0; kk < 256; kk += 32) {
      FragBF a, bb;
      a.q[0]  = *reinterpret_cast<const uint4*>(arow + kk + hi * 8);        // ds_load_b128
      a.q[1]  = *reinterpret_cast<const uint4*>(arow + kk + hi * 8 + 16);
      bb.q[0] = *reinterpret_cast<const uint4*>(brow + kk + hi * 8);        // global b128
      bb.q[1] = *reinterpret_cast<const uint4*>(brow + kk + hi * 8 + 16);
      acc = __builtin_amdgcn_wmma_f32_16x16x32_bf16(false, a.v, false, bb.v,
                                                    (short)0, acc, false, false);
    }
    #pragma unroll
    for (int j = 0; j < 8; ++j) {
      int m = m0 + hi * 8 + j;
      int n = n0 + l;
      float v = acc[j] + bias[n];
      if (do_tanh) v = tanhf(v);
      if (outLDS) outLDS[m * 256 + n] = f2bf(v);
      else        outG[(size_t)m * 256 + n] = v;
    }
  }
}

__global__ __launch_bounds__(1024)
void ode_kernel(float* __restrict__ h,
                const __bf16* __restrict__ w1t, const float* __restrict__ b1,
                const __bf16* __restrict__ w2t, const float* __restrict__ b2,
                const __bf16* __restrict__ w3t, const float* __restrict__ b3,
                float* __restrict__ kbuf)       // [6][256*256] f32
{
  __shared__ __align__(16) __bf16 lds0[256 * 256];   // 128 KB
  __shared__ __align__(16) __bf16 lds1[256 * 256];   // 128 KB
  const int tid  = threadIdx.x;
  const int lane = tid & 31;
  const int wave = tid >> 5;
  const float dt = 1.0f / 20.0f;

  for (int i = tid; i < 65536; i += 1024) lds0[i] = f2bf(h[i]);
  __syncthreads();

  for (int step = 0; step < 20; ++step) {
    for (int s = 0; s < 6; ++s) {
      float* kout = kbuf + (size_t)s * 65536;
      // f(arg): lds0 -> tanh -> lds1 -> tanh -> lds0 -> linear -> kout
      gemm256(lds0, w1t, b1, lds1, nullptr, true,  wave, lane);
      __syncthreads();
      gemm256(lds1, w2t, b2, lds0, nullptr, true,  wave, lane);
      __syncthreads();
      gemm256(lds0, w3t, b3, nullptr, kout, false, wave, lane);
      __threadfence();
      __syncthreads();
      // next-stage argument (or final state update at s == 5)
      for (int i = tid; i < 65536; i += 1024) {
        float a = 0.f;
        #pragma unroll
        for (int j = 0; j < 6; ++j)
          if (j <= s) a += CO[s][j] * kbuf[(size_t)j * 65536 + i];
        float v = h[i] + dt * a;
        if (s == 5) h[i] = v;
        lds0[i] = f2bf(v);
      }
      if (s == 5) __threadfence();
      __syncthreads();
    }
  }
}

// ---------------------------------------------------------------------------
// Classifier: logits[256][10] = h @ cls_w + cls_b  (tiny; plain VALU)
// ---------------------------------------------------------------------------
__global__ void cls_kernel(const float* __restrict__ h, const float* __restrict__ w,
                           const float* __restrict__ b, float* __restrict__ out)
{
  int idx = blockIdx.x * 256 + threadIdx.x;
  if (idx >= 2560) return;
  int bb = idx / 10;
  int j  = idx - bb * 10;
  float s = b[j];
  for (int k = 0; k < 256; ++k) s += h[bb * 256 + k] * w[k * 10 + j];
  out[idx] = s;
}

// ---------------------------------------------------------------------------
// Host orchestration
// ---------------------------------------------------------------------------
static inline int nb(int total) { return (total + 255) / 256; }

extern "C" void kernel_launch(void* const* d_in, const int* in_sizes, int n_in,
                              void* d_out, int out_size, void* d_ws, size_t ws_size,
                              hipStream_t stream)
{
  (void)in_sizes; (void)n_in; (void)out_size; (void)ws_size;
  const float* x = (const float*)d_in[0];
  const float *convw[5], *bng[5], *bnb[5];
  for (int i = 0; i < 5; ++i) {
    convw[i] = (const float*)d_in[4 * i + 1];
    bng[i]   = (const float*)d_in[4 * i + 3];
    bnb[i]   = (const float*)d_in[4 * i + 4];
  }
  const float* proj_w = (const float*)d_in[21];
  const float* proj_b = (const float*)d_in[22];
  const float* ode_w[3] = { (const float*)d_in[23], (const float*)d_in[25], (const float*)d_in[27] };
  const float* ode_b[3] = { (const float*)d_in[24], (const float*)d_in[26], (const float*)d_in[28] };
  const float* cls_w = (const float*)d_in[29];
  const float* cls_b = (const float*)d_in[30];
  float* out = (float*)d_out;

  // ---- workspace carve-up ----
  char* p = (char*)d_ws;
  auto alloc = [&](size_t bytes) { char* r = p; p += (bytes + 255) & ~(size_t)255; return r; };
  __bf16* bufA  = (__bf16*)alloc((size_t)64 * 262144 * 2);   // 32 MB
  __bf16* bufB  = (__bf16*)alloc((size_t)64 * 262144 * 2);   // 32 MB
  __bf16* in_bf = (__bf16*)alloc((size_t)3 * BATCH * 1024 * 2);
  const int Cin[5]  = { 3, 64, 64, 128, 128 };
  const int Cout[5] = { 64, 64, 128, 128, 256 };
  const int Kp16[5] = { 64, 1024, 1024, 2048, 2048 };   // ceil_even(Cin)*16
  const int Hs[5]   = { 32, 32, 16, 16, 8 };
  __bf16* wbf[5];
  for (int i = 0; i < 5; ++i) wbf[i] = (__bf16*)alloc((size_t)Cout[i] * Kp16[i] * 2);
  __bf16* projT = (__bf16*)alloc((size_t)256 * 4096 * 2);
  __bf16* odeT[3];
  for (int i = 0; i < 3; ++i) odeT[i] = (__bf16*)alloc((size_t)256 * 256 * 2);
  float*  h     = (float*)alloc((size_t)256 * 256 * 4);
  float*  kbuf  = (float*)alloc((size_t)6 * 65536 * 4);
  float*  stats = (float*)alloc((size_t)256 * 2 * 4);
  __bf16* zpad  = (__bf16*)alloc((size_t)128 * 2);

  // ---- prep: conversions / transposes ----
  zero_kernel<<<1, 128, 0, stream>>>(zpad);
  cvt_x_kernel<<<nb(3 * BATCH * 1024), 256, 0, stream>>>(x, in_bf);
  for (int i = 0; i < 5; ++i) {
    int total = Cout[i] * Kp16[i];
    cvt_convw_kernel<<<nb(total), 256, 0, stream>>>(convw[i], wbf[i], Cin[i], Kp16[i], total);
  }
  cvt_T_kernel<<<nb(4096 * 256), 256, 0, stream>>>(proj_w, projT, 4096, 256);
  for (int i = 0; i < 3; ++i)
    cvt_T_kernel<<<nb(256 * 256), 256, 0, stream>>>(ode_w[i], odeT[i], 256, 256);

  // ---- conv pipeline (ping-pong: src -> pp[d], optional pool -> pp[1-d]) ----
  __bf16* pp[2] = { bufA, bufB };
  const __bf16* src = in_bf;
  int d = 0;
  for (int i = 0; i < 5; ++i) {
    int H = Hs[i], W = Hs[i];
    int Npix = BATCH * H * W;
    __bf16* dst = pp[d];
    dim3 grid(Npix / 128, 1);
    switch (Cout[i] / 16) {
      case 4:  conv_gemm_kernel<4><<<grid, 256, 0, stream>>>(src, wbf[i], dst, zpad,
                 Cin[i], H, W, Kp16[i], Npix); break;
      case 8:  conv_gemm_kernel<8><<<grid, 256, 0, stream>>>(src, wbf[i], dst, zpad,
                 Cin[i], H, W, Kp16[i], Npix); break;
      default: conv_gemm_kernel<16><<<grid, 256, 0, stream>>>(src, wbf[i], dst, zpad,
                 Cin[i], H, W, Kp16[i], Npix); break;
    }
    bn_stats_kernel<<<Cout[i], 1024, 0, stream>>>(dst, stats, Npix);
    int total = Cout[i] * Npix;
    bn_apply_kernel<<<nb(total), 256, 0, stream>>>(dst, stats, bng[i], bnb[i], Npix, total);
    bool pool = (i == 1) || (i == 3) || (i == 4);
    if (pool) {
      int ptotal = Cout[i] * BATCH * (H / 2) * (W / 2);
      maxpool_kernel<<<nb(ptotal), 256, 0, stream>>>(dst, pp[1 - d], H, W, ptotal);
      src = pp[1 - d];             // d unchanged: pp[d] is dead, reused next conv
    } else {
      src = pp[d];
      d = 1 - d;
    }
  }
  // After conv5+bn5+pool3: src = pooled activations [256][256][4][4] bf16

  // ---- projection: feats @ proj_w + b -> h [256][256] f32 ----
  {
    dim3 grid(2, 16);
    proj_gemm_kernel<<<grid, 256, 0, stream>>>(src, projT, proj_b, h);
  }

  // ---- persistent dopri5 integrator (one WGP, 32 waves, 360 WMMA GEMMs) ----
  ode_kernel<<<1, 1024, 0, stream>>>(h, odeT[0], ode_b[0], odeT[1], ode_b[1],
                                     odeT[2], ode_b[2], kbuf);

  // ---- classifier ----
  cls_kernel<<<nb(2560), 256, 0, stream>>>(h, cls_w, cls_b, out);
}
